// ContextualAttention_66125316489940
// MI455X (gfx1250) — matrix operand order repaired and checked by reading, not compiled, not run on hardware
//
#include <hip/hip_runtime.h>
#include <hip/hip_bf16.h>
#include <cstdint>

typedef __attribute__((ext_vector_type(16))) __bf16 v16bf;
typedef __attribute__((ext_vector_type(8)))  float  v8f;

#define LSZ 2304   // 48*48 patch positions
#define K1  1152   // 128*3*3
#define N2  2048   // 128*4*4
#define CH  128
#define HW  96

// ---------------------------------------------------------------------------
// Prep 1: 3x3 patches of x2s (=x2[::2,::2]) -> raw P (bf16), normalized Wn
// (bf16, norm in fp32), and mask gate mm. One block per patch l, 128 threads.
// ---------------------------------------------------------------------------
__global__ void k_prep_patches(const float* __restrict__ x2,
                               const float* __restrict__ mask,
                               __bf16* __restrict__ P, __bf16* __restrict__ Wn,
                               float* __restrict__ mm, int b)
{
    __shared__ float red[128];
    int l = blockIdx.x;
    int i = l / 48, j = l % 48;
    int c = threadIdx.x;
    const float* xb = x2 + ((size_t)(b * CH + c)) * HW * HW;
    float vals[9];
    float ss = 0.f;
#pragma unroll
    for (int di = 0; di < 3; ++di) {
        int y = i - 1 + di;
#pragma unroll
        for (int dj = 0; dj < 3; ++dj) {
            int x = j - 1 + dj;
            float v = 0.f;
            if ((unsigned)y < 48u && (unsigned)x < 48u) v = xb[(2 * y) * HW + 2 * x];
            vals[di * 3 + dj] = v;
            ss += v * v;
        }
    }
    red[c] = ss;
    __syncthreads();
    for (int s = 64; s > 0; s >>= 1) { if (c < s) red[c] += red[c + s]; __syncthreads(); }
    float inv = 1.f / fmaxf(sqrtf(red[0]), 1e-4f);
    __bf16* Pr = P  + (size_t)l * K1 + c * 9;
    __bf16* Wr = Wn + (size_t)l * K1 + c * 9;
#pragma unroll
    for (int t = 0; t < 9; ++t) { Pr[t] = (__bf16)vals[t]; Wr[t] = (__bf16)(vals[t] * inv); }
    if (c == 0) {
        const float* mb = mask + (size_t)b * 48 * 48;
        float s2 = 0.f;
        for (int di = 0; di < 3; ++di) { int y = i - 1 + di;
            for (int dj = 0; dj < 3; ++dj) { int x = j - 1 + dj;
                if ((unsigned)y < 48u && (unsigned)x < 48u) s2 += mb[y * 48 + x]; } }
        mm[l] = (s2 * (1.f / 9.f) == 0.f) ? 1.f : 0.f;
    }
}

// ---------------------------------------------------------------------------
// Prep 2: rawWT[m][l] = x1[c, 2*pi-1+ki, 2*pj-1+kj], m=(ki*4+kj)*128+c (bf16)
// ---------------------------------------------------------------------------
__global__ void k_rawwt(const float* __restrict__ x1, __bf16* __restrict__ RW, int b)
{
    int idx = blockIdx.x * blockDim.x + threadIdx.x;   // 2048*2304 threads
    int m = idx / LSZ;
    int l = idx % LSZ;
    if (m >= N2) return;
    int kk = m >> 7, c = m & 127;
    int ki = kk >> 2, kj = kk & 3;
    int pi = l / 48, pj = l % 48;
    int u = 2 * pi - 1 + ki, v = 2 * pj - 1 + kj;
    float val = 0.f;
    if ((unsigned)u < 96u && (unsigned)v < 96u)
        val = x1[(((size_t)b * CH + c) * HW + u) * HW + v];
    RW[(size_t)m * LSZ + l] = (__bf16)val;
}

// ---------------------------------------------------------------------------
// bf16 WMMA GEMM: C[M x N] = A[M x K] * Bt[N x K]^T, f32 accumulate.
// 128x128 block tile, 8 waves (wave32), each wave 64x32 (4x2 fragments).
// K stepped by 32. Global->LDS staging uses gfx1250 async direct-to-LDS
// loads (ASYNCcnt) with double-buffered tiles so the next tile streams in
// while the current tile feeds the WMMAs. All dims divide tile sizes.
// ---------------------------------------------------------------------------
#define ASYNC_LDS_B128(ldsoff, gaddr)                                     \
    asm volatile("global_load_async_to_lds_b128 %0, %1, off"              \
                 :: "v"(ldsoff), "v"(gaddr) : "memory")

__global__ __launch_bounds__(256) void k_gemm_bf16(
    const __bf16* __restrict__ A, const __bf16* __restrict__ Bt,
    float* __restrict__ Cm, int Ndim, int Kdim)
{
    // [A buf0 | A buf1 | B buf0 | B buf1], each 128x32 bf16 = 8 KB
    __shared__ __align__(16) __bf16 smem[4 * 128 * 32];
    int tid = threadIdx.x;
    int lane = tid & 31, w = tid >> 5;
    int wm = w >> 2, wn = w & 3;          // 2 x 4 wave grid
    int rowA0 = blockIdx.y * 128;
    int rowB0 = blockIdx.x * 128;
    int r0 = tid >> 2, q = tid & 3;       // tile fill mapping: 2 rows/thread
    int laneM = lane & 15;
    int kb  = (lane >> 4) * 8;            // A frag K base
    int kb2 = (lane >> 4) * 16;           // unused; B frags use kb
    union F { uint4 q2[2]; v16bf v; };
    v8f acc[4][2] = {};

    // LDS byte offsets (low 32 bits of the generic pointer = LDS offset)
    uint32_t sBase = (uint32_t)(uintptr_t)&smem[0];
    uint32_t offA0 = sBase + (uint32_t)(r0 * 64 + q * 16);
    uint32_t offA1 = offA0 + 64 * 64;             // +64 rows
    uint32_t offB0 = offA0 + 16384;
    uint32_t offB1 = offA1 + 16384;

    // per-thread global source addresses (advance 64 B per K-step)
    uint64_t gA0 = (uint64_t)(uintptr_t)(A  + (size_t)(rowA0 + r0)      * Kdim + q * 8);
    uint64_t gA1 = (uint64_t)(uintptr_t)(A  + (size_t)(rowA0 + r0 + 64) * Kdim + q * 8);
    uint64_t gB0 = (uint64_t)(uintptr_t)(Bt + (size_t)(rowB0 + r0)      * Kdim + q * 8);
    uint64_t gB1 = (uint64_t)(uintptr_t)(Bt + (size_t)(rowB0 + r0 + 64) * Kdim + q * 8);

    // prologue: stream tile 0 into buffer 0
    ASYNC_LDS_B128(offA0, gA0);
    ASYNC_LDS_B128(offA1, gA1);
    ASYNC_LDS_B128(offB0, gB0);
    ASYNC_LDS_B128(offB1, gB1);
    gA0 += 64; gA1 += 64; gB0 += 64; gB1 += 64;
    asm volatile("s_wait_asynccnt 0x0" ::: "memory");
    __syncthreads();

    int nsteps = Kdim >> 5;
    uint32_t buf = 0;
    for (int s = 0; s < nsteps; ++s) {
        if (s + 1 < nsteps) {
            uint32_t bo = (buf ^ 1u) * 8192u;     // next buffer byte offset
            ASYNC_LDS_B128(offA0 + bo, gA0);
            ASYNC_LDS_B128(offA1 + bo, gA1);
            ASYNC_LDS_B128(offB0 + bo, gB0);
            ASYNC_LDS_B128(offB1 + bo, gB1);
            gA0 += 64; gA1 += 64; gB0 += 64; gB1 += 64;
        }
        const __bf16* As = smem + buf * 4096;
        const __bf16* Bs = smem + 8192 + buf * 4096;
        F af[4], bfr[2];
#pragma unroll
        for (int fm = 0; fm < 4; ++fm) {
            int Mr = wm * 64 + fm * 16 + laneM;
            af[fm].q2[0] = *(const uint4*)&As[Mr * 32 + kb];
            af[fm].q2[1] = *(const uint4*)&As[Mr * 32 + kb + 16];
        }
#pragma unroll
        for (int fn = 0; fn < 2; ++fn) {
            int Nr = wn * 32 + fn * 16 + laneM;
            bfr[fn].q2[0] = *(const uint4*)&Bs[Nr * 32 + kb];
            bfr[fn].q2[1] = *(const uint4*)&Bs[Nr * 32 + kb + 16];
        }
#pragma unroll
        for (int fm = 0; fm < 4; ++fm)
#pragma unroll
            for (int fn = 0; fn < 2; ++fn)
                acc[fm][fn] = __builtin_amdgcn_wmma_f32_16x16x32_bf16(
                    false, af[fm].v, false, bfr[fn].v,
                    (short)0, acc[fm][fn], false, false);
        if (s + 1 < nsteps)
            asm volatile("s_wait_asynccnt 0x0" ::: "memory");
        __syncthreads();
        buf ^= 1u;
    }
#pragma unroll
    for (int fm = 0; fm < 4; ++fm) {
        int mBase = rowA0 + wm * 64 + fm * 16 + (lane >> 4) * 8;
#pragma unroll
        for (int fn = 0; fn < 2; ++fn) {
            int nI = rowB0 + wn * 32 + fn * 16 + laneM;
#pragma unroll
            for (int r = 0; r < 8; ++r)
                Cm[(size_t)(mBase + r) * Ndim + nI] = acc[fm][fn][r];
        }
    }
}

// ---------------------------------------------------------------------------
// fuse -> transpose -> fuse composed directly: Zt[p][l] = sum of <=9 S terms.
// phi(t) = (t%48)*48 + t/48 (the 4D transpose on the flattened index).
// ---------------------------------------------------------------------------
__global__ void k_fuse(const float* __restrict__ S, float* __restrict__ Zt)
{
    int l = blockIdx.x * 16 + threadIdx.x;
    int p = blockIdx.y * 16 + threadIdx.y;
    int pl = (l % 48) * 48 + l / 48;
    int pp = (p % 48) * 48 + p / 48;
    float sum = 0.f;
#pragma unroll
    for (int d = -1; d <= 1; ++d) {
        int il = pl + d, jp = pp + d;
        if ((unsigned)il >= (unsigned)LSZ || (unsigned)jp >= (unsigned)LSZ) continue;
        int a  = (il % 48) * 48 + il / 48;
        int bb = (jp % 48) * 48 + jp / 48;
#pragma unroll
        for (int e = -1; e <= 1; ++e) {
            int rr = a + e, cc = bb + e;
            if ((unsigned)rr < (unsigned)LSZ && (unsigned)cc < (unsigned)LSZ)
                sum += S[(size_t)rr * LSZ + cc];
        }
    }
    Zt[(size_t)p * LSZ + l] = sum;
}

// ---------------------------------------------------------------------------
// Softmax over l for each position p; apply mask gate before (scale 10) and
// after (clamp 1e-8); emit bf16 for the paint GEMM. One block per p.
// ---------------------------------------------------------------------------
__global__ void k_softmax(const float* __restrict__ Zt, const float* __restrict__ mm,
                          __bf16* __restrict__ Y)
{
    __shared__ float red[256];
    int p = blockIdx.x, t = threadIdx.x;
    const float* row = Zt + (size_t)p * LSZ;
    float mx = -3.4e38f;
    for (int l = t; l < LSZ; l += 256) { float v = row[l] * mm[l] * 10.f; mx = fmaxf(mx, v); }
    red[t] = mx; __syncthreads();
    for (int s = 128; s > 0; s >>= 1) { if (t < s) red[t] = fmaxf(red[t], red[t + s]); __syncthreads(); }
    mx = red[0]; __syncthreads();
    float sum = 0.f;
    for (int l = t; l < LSZ; l += 256) { float v = row[l] * mm[l] * 10.f; sum += __expf(v - mx); }
    red[t] = sum; __syncthreads();
    for (int s = 128; s > 0; s >>= 1) { if (t < s) red[t] += red[t + s]; __syncthreads(); }
    float inv = 1.f / red[0];
    for (int l = t; l < LSZ; l += 256) {
        float v = row[l] * mm[l] * 10.f;
        float y = __expf(v - mx) * inv;
        y = fmaxf(y * mm[l], 1e-8f);
        Y[(size_t)p * LSZ + l] = (__bf16)y;
    }
}

// ---------------------------------------------------------------------------
// Stride-2 overlap-add of per-position 4x4x128 painted patches; /4.
// One block per (u,v) output pixel, 128 threads over channels.
// ---------------------------------------------------------------------------
__global__ void k_scatter(const float* __restrict__ OP, float* __restrict__ out, int b)
{
    int uv = blockIdx.x;
    int u = uv / 96, v = uv % 96;
    int c = threadIdx.x;
    float sum = 0.f;
    int k0u = (u + 1) & 1, k0v = (v + 1) & 1;
#pragma unroll
    for (int ti = 0; ti < 2; ++ti) {
        int ki = k0u + 2 * ti, ii = (u + 1 - ki) >> 1;
        if ((unsigned)ii >= 48u) continue;
#pragma unroll
        for (int tj = 0; tj < 2; ++tj) {
            int kj = k0v + 2 * tj, jj = (v + 1 - kj) >> 1;
            if ((unsigned)jj >= 48u) continue;
            sum += OP[(size_t)(ii * 48 + jj) * N2 + ((ki * 4 + kj) << 7) + c];
        }
    }
    out[(((size_t)b * CH + c) * HW + u) * HW + v] = 0.25f * sum;
}

// ---------------------------------------------------------------------------
extern "C" void kernel_launch(void* const* d_in, const int* in_sizes, int n_in,
                              void* d_out, int out_size, void* d_ws, size_t ws_size,
                              hipStream_t stream)
{
    (void)in_sizes; (void)n_in; (void)out_size; (void)ws_size;
    const float* x1   = (const float*)d_in[0];
    const float* x2   = (const float*)d_in[1];
    const float* mask = (const float*)d_in[2];
    float* out = (float*)d_out;

    char* ws = (char*)d_ws;
    size_t off = 0;
    auto alloc = [&](size_t bytes) -> void* {
        void* p = ws + off;
        off = (off + bytes + 255) & ~(size_t)255;
        return p;
    };
    __bf16* P   = (__bf16*)alloc((size_t)LSZ * K1 * 2);   // raw patches
    __bf16* Wn  = (__bf16*)alloc((size_t)LSZ * K1 * 2);   // normalized patches
    __bf16* RW  = (__bf16*)alloc((size_t)N2  * LSZ * 2);  // rawW transposed
    float*  mm  = (float*) alloc((size_t)LSZ * 4);        // mask gate
    float*  S   = (float*) alloc((size_t)LSZ * LSZ * 4);  // scores (reused as Ybf)
    float*  Zt  = (float*) alloc((size_t)LSZ * LSZ * 4);  // fused scores (reused as OP)
    __bf16* Ybf = (__bf16*)S;   // softmax output overlays S (S dead by then)
    float*  OP  = Zt;           // paint GEMM output overlays Zt (Zt dead by then)

    for (int b = 0; b < 2; ++b) {
        k_prep_patches<<<LSZ, 128, 0, stream>>>(x2, mask, P, Wn, mm, b);
        k_rawwt<<<(N2 * LSZ) / 256, 256, 0, stream>>>(x1, RW, b);
        // scores: S = Wn @ P^T  (2304 x 2304 x 1152)
        k_gemm_bf16<<<dim3(LSZ / 128, LSZ / 128), 256, 0, stream>>>(Wn, P, S, LSZ, K1);
        k_fuse<<<dim3(LSZ / 16, LSZ / 16), dim3(16, 16), 0, stream>>>(S, Zt);
        k_softmax<<<LSZ, 256, 0, stream>>>(Zt, mm, Ybf);
        // paint: OP = Y^T @ rawW  (2304 x 2048 x 2304)
        k_gemm_bf16<<<dim3(N2 / 128, LSZ / 128), 256, 0, stream>>>(Ybf, RW, OP, N2, LSZ);
        k_scatter<<<HW * HW, 128, 0, stream>>>(OP, out, b);
    }
}
